// RelativePositionalMultiHeadAttention_45947560132902
// MI455X (gfx1250) — matrix operand order repaired and checked by reading, not compiled, not run on hardware
//
#include <hip/hip_runtime.h>
#include <hip/hip_bf16.h>

// ---------------------------------------------------------------------------
// CDNA5 (gfx1250) relative-positional MHA, bf16 WMMA pipeline.
//  - specialized GEMM instantiations (no runtime flag branches)
//  - double-buffered LDS; async global->LDS copies (ASYNCcnt) overlap WMMA
//  - hardware f32->bf16 conversion
// ---------------------------------------------------------------------------

typedef __attribute__((ext_vector_type(16))) __bf16 v16bf;
typedef __attribute__((ext_vector_type(8)))  float  v8f;

#define LDT 40   // LDS row stride in halves (32 data + 8 pad -> conflict-free)
#define TPB 256

#ifdef __HIP_DEVICE_COMPILE__
#if defined(__has_builtin)
#if __has_builtin(__builtin_amdgcn_global_load_async_to_lds_b128)
#define HAVE_ASYNC_LDS 1
#endif
#endif
#endif
#ifndef HAVE_ASYNC_LDS
#define HAVE_ASYNC_LDS 0
#endif

#if HAVE_ASYNC_LDS
typedef int v4i_vec __attribute__((vector_size(16)));          // int __vector(4)
typedef __attribute__((address_space(1))) v4i_vec gbl_v4i;     // global int4
typedef __attribute__((address_space(3))) v4i_vec lds_v4i;     // LDS int4
#endif

static __device__ inline void wait_async0() {
#if HAVE_ASYNC_LDS
#if __has_builtin(__builtin_amdgcn_s_wait_asynccnt)
    __builtin_amdgcn_s_wait_asynccnt(0);
#else
    asm volatile("s_wait_asynccnt 0" ::: "memory");
#endif
#endif
}

static __device__ inline unsigned short f2bf(float f) {
#ifdef __HIP_DEVICE_COMPILE__
    // native convert: lowers to v_cvt_pk_bf16_f32-class ops (RNE)
    __bf16 h = (__bf16)f;
    return __builtin_bit_cast(unsigned short, h);
#else
    unsigned int u = __float_as_uint(f);
    unsigned int r = (u + 0x7FFFu + ((u >> 16) & 1u)) >> 16;  // RNE
    return (unsigned short)r;
#endif
}
static __device__ inline float bf2f(unsigned short u) {
    return __uint_as_float(((unsigned int)u) << 16);
}

union FragCvt {
    struct { uint4 lo, hi; } q;
    v16bf v;
};
static __device__ inline v16bf make_frag(uint4 lo, uint4 hi) {
    FragCvt fc;
    fc.q.lo = lo; fc.q.hi = hi;
    return fc.v;
}

struct GemmArgs {
    // A: [M x K], row stride a_rs. batch offset = (z>>4)*a_bs_b + (z&15)*a_bs_h
    const void* A;  long long a_bs_b, a_bs_h; int a_rs;
    // B: normally [N x K] (row = output column, contiguous K). If B_KXN, memory
    // is [K x N] with row stride b_rs; loader transposes into LDS.
    const void* Bm; long long b_bs_b, b_bs_h; int b_rs;
    // out0: [M x N], row stride o_rs; fp32 or bf16 per template.
    void* out0;     long long o_bs_b, o_bs_h; int o_rs;
    const float* bias0a; const float* bias0b;
    // optional second bf16 output (same layout), different bias pair
    unsigned short* out1; const float* bias1a; const float* bias1b;
    int M, N, K;
};

// C = A * B^T (+bias), 128x128 tile/workgroup, BK=32, bf16 WMMA, f32 accum.
// Double-buffered LDS: next tile's async copies / global loads are issued
// before the current tile's WMMA block and committed/awaited after it.
template <int A_BF16, int B_BF16, int B_KXN, int DUAL, int OUT_F32, int NB0>
__global__ __launch_bounds__(TPB) void gemm_kern(GemmArgs g) {
    __shared__ unsigned short sA[2][128 * LDT];
    __shared__ unsigned short sB[2][128 * LDT];

    constexpr bool ASYNC_A = HAVE_ASYNC_LDS && A_BF16;
    constexpr bool ASYNC_B = HAVE_ASYNC_LDS && B_BF16 && !B_KXN;

    const int tid  = threadIdx.x;
    const int lane = tid & 31;
    const int wid  = tid >> 5;
    const int wm   = wid & 3;    // 4 wave-rows  * 32 rows
    const int wn   = wid >> 2;   // 2 wave-cols  * 64 cols
    const int m0   = blockIdx.y * 128;
    const int n0   = blockIdx.x * 128;
    const int z    = blockIdx.z;
    const int zb   = z >> 4, zh = z & 15;

    const long long aoff = (long long)zb * g.a_bs_b + (long long)zh * g.a_bs_h;
    const long long boff = (long long)zb * g.b_bs_b + (long long)zh * g.b_bs_h;
    const long long ooff = (long long)zb * g.o_bs_b + (long long)zh * g.o_bs_h;

    const v8f vzero = {0.f, 0.f, 0.f, 0.f, 0.f, 0.f, 0.f, 0.f};
    v8f acc[2][4];
#pragma unroll
    for (int i = 0; i < 2; ++i)
#pragma unroll
        for (int j = 0; j < 4; ++j) acc[i][j] = vzero;

    const int lh = lane >> 4;   // half-wave select
    const int lr = lane & 15;

    // staging registers (only the variant selected by the template survives)
    float4  ra[4];    // A fp32 path
    float4  rbf[4];   // B fp32 path
    ushort4 rau[4];   // A bf16 non-async fallback
    ushort4 rbu[4];   // B bf16 non-async fallback
    unsigned short rbk[16];  // B kxn transpose path

    // ---- A staging -------------------------------------------------------
    auto stageA_issue = [&](int kk, int b) {
        if constexpr (ASYNC_A) {
#pragma unroll
            for (int it = 0; it < 2; ++it) {
                int lin = it * TPB + tid;
                int row = lin >> 2;
                int cg  = (lin & 3) << 3;   // 8 halves = 16B
                const unsigned short* src = (const unsigned short*)g.A + aoff +
                    (long long)(m0 + row) * g.a_rs + kk + cg;
#if HAVE_ASYNC_LDS
                __builtin_amdgcn_global_load_async_to_lds_b128(
                    (gbl_v4i*)(void*)src, (lds_v4i*)&sA[b][row * LDT + cg], 0, 0);
#endif
            }
        } else if constexpr (A_BF16) {
#pragma unroll
            for (int it = 0; it < 4; ++it) {
                int lin = it * TPB + tid;
                int row = lin >> 3;
                int cg  = (lin & 7) << 2;
                const unsigned short* src = (const unsigned short*)g.A + aoff +
                    (long long)(m0 + row) * g.a_rs + kk + cg;
                rau[it].x = src[0]; rau[it].y = src[1];
                rau[it].z = src[2]; rau[it].w = src[3];
            }
        } else {
#pragma unroll
            for (int it = 0; it < 4; ++it) {
                int lin = it * TPB + tid;
                int row = lin >> 3;
                int cg  = (lin & 7) << 2;
                const float* src = (const float*)g.A + aoff +
                    (long long)(m0 + row) * g.a_rs + kk + cg;
                ra[it] = *(const float4*)src;
            }
        }
    };
    auto stageA_commit = [&](int b) {
        if constexpr (ASYNC_A) {
            // nothing: async engine writes LDS directly
        } else if constexpr (A_BF16) {
#pragma unroll
            for (int it = 0; it < 4; ++it) {
                int lin = it * TPB + tid;
                int row = lin >> 3;
                int cg  = (lin & 7) << 2;
                *(ushort4*)&sA[b][row * LDT + cg] = rau[it];
            }
        } else {
#pragma unroll
            for (int it = 0; it < 4; ++it) {
                int lin = it * TPB + tid;
                int row = lin >> 3;
                int cg  = (lin & 7) << 2;
                ushort4 st;
                st.x = f2bf(ra[it].x); st.y = f2bf(ra[it].y);
                st.z = f2bf(ra[it].z); st.w = f2bf(ra[it].w);
                *(ushort4*)&sA[b][row * LDT + cg] = st;
            }
        }
    };

    // ---- B staging -------------------------------------------------------
    auto stageB_issue = [&](int kk, int b) {
        if constexpr (B_KXN) {
            const unsigned short* Bb = (const unsigned short*)g.Bm + boff;
#pragma unroll
            for (int it = 0; it < 4; ++it) {
                int lin  = it * TPB + tid;
                int kloc = lin >> 5;
                int ng   = (lin & 31) << 2;
                const unsigned short* src = Bb + (long long)(kk + kloc) * g.b_rs + n0 + ng;
#pragma unroll
                for (int jj = 0; jj < 4; ++jj)
                    rbk[it * 4 + jj] = (n0 + ng + jj < g.N) ? src[jj] : (unsigned short)0;
            }
        } else if constexpr (ASYNC_B) {
#pragma unroll
            for (int it = 0; it < 2; ++it) {
                int lin = it * TPB + tid;
                int row = lin >> 2;
                int cg  = (lin & 3) << 3;
                const unsigned short* src = (const unsigned short*)g.Bm + boff +
                    (long long)(n0 + row) * g.b_rs + kk + cg;
#if HAVE_ASYNC_LDS
                __builtin_amdgcn_global_load_async_to_lds_b128(
                    (gbl_v4i*)(void*)src, (lds_v4i*)&sB[b][row * LDT + cg], 0, 0);
#endif
            }
        } else if constexpr (B_BF16) {
#pragma unroll
            for (int it = 0; it < 4; ++it) {
                int lin = it * TPB + tid;
                int row = lin >> 3;
                int cg  = (lin & 7) << 2;
                const unsigned short* src = (const unsigned short*)g.Bm + boff +
                    (long long)(n0 + row) * g.b_rs + kk + cg;
                rbu[it].x = src[0]; rbu[it].y = src[1];
                rbu[it].z = src[2]; rbu[it].w = src[3];
            }
        } else {
#pragma unroll
            for (int it = 0; it < 4; ++it) {
                int lin = it * TPB + tid;
                int row = lin >> 3;
                int cg  = (lin & 7) << 2;
                const float* src = (const float*)g.Bm + boff +
                    (long long)(n0 + row) * g.b_rs + kk + cg;
                rbf[it] = *(const float4*)src;
            }
        }
    };
    auto stageB_commit = [&](int b) {
        if constexpr (B_KXN) {
#pragma unroll
            for (int it = 0; it < 4; ++it) {
                int lin  = it * TPB + tid;
                int kloc = lin >> 5;
                int ng   = (lin & 31) << 2;
#pragma unroll
                for (int jj = 0; jj < 4; ++jj)
                    sB[b][(ng + jj) * LDT + kloc] = rbk[it * 4 + jj];
            }
        } else if constexpr (ASYNC_B) {
            // nothing
        } else if constexpr (B_BF16) {
#pragma unroll
            for (int it = 0; it < 4; ++it) {
                int lin = it * TPB + tid;
                int row = lin >> 3;
                int cg  = (lin & 7) << 2;
                *(ushort4*)&sB[b][row * LDT + cg] = rbu[it];
            }
        } else {
#pragma unroll
            for (int it = 0; it < 4; ++it) {
                int lin = it * TPB + tid;
                int row = lin >> 3;
                int cg  = (lin & 7) << 2;
                ushort4 st;
                st.x = f2bf(rbf[it].x); st.y = f2bf(rbf[it].y);
                st.z = f2bf(rbf[it].z); st.w = f2bf(rbf[it].w);
                *(ushort4*)&sB[b][row * LDT + cg] = st;
            }
        }
    };

    // ---- WMMA block: fragments per ISA 7.12.2 layouts --------------------
    auto compute = [&](int b) {
        v16bf af[2], bfrag[4];
#pragma unroll
        for (int i = 0; i < 2; ++i) {
            int row = wm * 32 + i * 16 + lr;
            const uint4* p0 = (const uint4*)&sA[b][row * LDT + lh * 8];       // K 0-7 / 8-15
            const uint4* p1 = (const uint4*)&sA[b][row * LDT + 16 + lh * 8];  // K 16-23 / 24-31
            af[i] = make_frag(p0[0], p1[0]);
        }
#pragma unroll
        for (int j = 0; j < 4; ++j) {
            int col = wn * 64 + j * 16 + lr;
            const uint4* q0 = (const uint4*)&sB[b][col * LDT + lh * 8];       // K 0-7 / 8-15
            const uint4* q1 = (const uint4*)&sB[b][col * LDT + 16 + lh * 8];  // K 16-23 / 24-31
            bfrag[j] = make_frag(q0[0], q1[0]);
        }
#pragma unroll
        for (int i = 0; i < 2; ++i)
#pragma unroll
            for (int j = 0; j < 4; ++j)
                acc[i][j] = __builtin_amdgcn_wmma_f32_16x16x32_bf16(
                    false, af[i], false, bfrag[j], (short)0, acc[i][j], false, false);
    };

    // ---- software pipeline ----------------------------------------------
    int buf = 0;
    stageA_issue(0, 0);
    stageB_issue(0, 0);
    stageA_commit(0);
    stageB_commit(0);
    if constexpr (ASYNC_A || ASYNC_B) wait_async0();
    __syncthreads();

    for (int kk = 0; kk < g.K; kk += 32) {
        const int  nkk     = kk + 32;
        const bool hasNext = nkk < g.K;
        if (hasNext) {                 // issue next tile (async overlaps WMMA)
            stageA_issue(nkk, buf ^ 1);
            stageB_issue(nkk, buf ^ 1);
        }
        compute(buf);
        if (hasNext) {                 // register->LDS commits after compute
            stageA_commit(buf ^ 1);
            stageB_commit(buf ^ 1);
        }
        if constexpr (ASYNC_A || ASYNC_B) wait_async0();
        __syncthreads();
        buf ^= 1;
    }

    // ---- epilogue: C layout VGPR e -> m = (lane>>4)*8+e, n = lane&15 ----
#pragma unroll
    for (int i = 0; i < 2; ++i) {
        int mbase = m0 + wm * 32 + i * 16 + lh * 8;
#pragma unroll
        for (int j = 0; j < 4; ++j) {
            int gn = n0 + wn * 64 + j * 16 + lr;
            if (gn >= g.N) continue;
            float b0 = 0.f, b1 = 0.f;
            if constexpr (NB0 >= 1) b0 += g.bias0a[gn];
            if constexpr (NB0 >= 2) b0 += g.bias0b[gn];
            if constexpr (DUAL) b1 = g.bias1a[gn] + g.bias1b[gn];
#pragma unroll
            for (int e = 0; e < 8; ++e) {
                float v = acc[i][j][e];
                long long idx = ooff + (long long)(mbase + e) * g.o_rs + gn;
                if constexpr (OUT_F32) ((float*)g.out0)[idx] = v + b0;
                else ((unsigned short*)g.out0)[idx] = f2bf(v + b0);
                if constexpr (DUAL) g.out1[idx] = f2bf(v + b1);
            }
        }
    }
}

// ---------------------------------------------------------------------------
// softmax over one score row with Transformer-XL relative shift applied as a
// gather from the unshifted pos matrix X:
//   shifted(i,j) = X[i, T-1-i+j]   if j <= i
//                = 0               if j == i+1
//                = X[i+1, j-i-2]   if j >= i+2
// ---------------------------------------------------------------------------
__global__ __launch_bounds__(128) void softmax_relshift(unsigned short* S,
                                                        const unsigned short* X) {
    const int T = 1024;
    const int i  = blockIdx.x;
    const long long bh = blockIdx.y;
    unsigned short* row = S + (bh * T + i) * T;
    const unsigned short* x0 = X + (bh * T + i) * T;
    const unsigned short* x1 = X + (bh * T + (i + 1)) * T;  // touched only when i <= T-3
    const int t = threadIdx.x;

    float s[8];
    float mx = -3.402823466e38f;
#pragma unroll
    for (int c = 0; c < 8; ++c) {
        int j = c * 128 + t;
        float cont = bf2f(row[j]);
        float pos;
        if (j <= i)            pos = bf2f(x0[T - 1 - i + j]);
        else if (j == i + 1)   pos = 0.f;
        else                   pos = bf2f(x1[j - i - 2]);
        s[c] = (cont + pos) * 0.125f;   // 1/sqrt(64)
        mx = fmaxf(mx, s[c]);
    }

    __shared__ float red[4];
#pragma unroll
    for (int o = 16; o > 0; o >>= 1) mx = fmaxf(mx, __shfl_xor(mx, o, 32));
    if ((t & 31) == 0) red[t >> 5] = mx;
    __syncthreads();
    mx = fmaxf(fmaxf(red[0], red[1]), fmaxf(red[2], red[3]));
    __syncthreads();

    float sum = 0.f;
#pragma unroll
    for (int c = 0; c < 8; ++c) { s[c] = __expf(s[c] - mx); sum += s[c]; }
#pragma unroll
    for (int o = 16; o > 0; o >>= 1) sum += __shfl_xor(sum, o, 32);
    if ((t & 31) == 0) red[t >> 5] = sum;
    __syncthreads();
    sum = red[0] + red[1] + red[2] + red[3];

    float inv = 1.f / sum;
#pragma unroll
    for (int c = 0; c < 8; ++c) row[c * 128 + t] = f2bf(s[c] * inv);
}

// ---------------------------------------------------------------------------
extern "C" void kernel_launch(void* const* d_in, const int* in_sizes, int n_in,
                              void* d_out, int out_size, void* d_ws, size_t ws_size,
                              hipStream_t stream) {
    const float* query   = (const float*)d_in[0];
    const float* key     = (const float*)d_in[1];
    const float* value   = (const float*)d_in[2];
    const float* pos_emb = (const float*)d_in[3];
    const float* Wq      = (const float*)d_in[4];
    const float* bq      = (const float*)d_in[5];
    const float* Wk      = (const float*)d_in[6];
    const float* bk      = (const float*)d_in[7];
    const float* Wv      = (const float*)d_in[8];
    const float* bv      = (const float*)d_in[9];
    const float* Wp      = (const float*)d_in[10];
    const float* pbu     = (const float*)d_in[11];  // [H, DK] == flat [1024]
    const float* pbv     = (const float*)d_in[12];
    const float* Wo      = (const float*)d_in[13];
    const float* bo      = (const float*)d_in[14];

    const int T = 1024, D = 1024;
    const long long MT = 4096;          // B*T
    const long long TT = (long long)T * T;

    char* ws = (char*)d_ws;
    size_t off = 0;
    auto take = [&](size_t bytes) -> void* {
        void* p = ws + off;
        off += (bytes + 255) & ~(size_t)255;
        return p;
    };
    unsigned short* qu  = (unsigned short*)take(MT * D * 2);
    unsigned short* qv  = (unsigned short*)take(MT * D * 2);
    unsigned short* kb  = (unsigned short*)take(MT * D * 2);
    unsigned short* vb  = (unsigned short*)take(MT * D * 2);
    unsigned short* pb  = (unsigned short*)take(MT * D * 2);
    unsigned short* S   = (unsigned short*)take(64 * TT * 2);   // content -> probs
    unsigned short* X   = (unsigned short*)take(64 * TT * 2);   // unshifted pos
    unsigned short* ctx = (unsigned short*)take(MT * D * 2);
    (void)ws_size; (void)in_sizes; (void)n_in; (void)out_size;

    dim3 blk(TPB);

    // 1) q projection -> q_u, q_v (dual bias)
    {
        GemmArgs a{};
        a.A = query; a.a_rs = 1024;
        a.Bm = Wq;   a.b_rs = 1024;
        a.out0 = qu; a.o_rs = 1024;
        a.bias0a = bq; a.bias0b = pbu;
        a.out1 = qv; a.bias1a = bq; a.bias1b = pbv;
        a.M = 4096; a.N = 1024; a.K = 1024;
        gemm_kern<0,0,0,1,0,2><<<dim3(8, 32, 1), blk, 0, stream>>>(a);
    }
    // 2) k  3) v  4) p projections
    {
        GemmArgs a{};
        a.A = key; a.a_rs = 1024; a.Bm = Wk; a.b_rs = 1024;
        a.out0 = kb; a.o_rs = 1024; a.bias0a = bk;
        a.M = 4096; a.N = 1024; a.K = 1024;
        gemm_kern<0,0,0,0,0,1><<<dim3(8, 32, 1), blk, 0, stream>>>(a);
    }
    {
        GemmArgs a{};
        a.A = value; a.a_rs = 1024; a.Bm = Wv; a.b_rs = 1024;
        a.out0 = vb; a.o_rs = 1024; a.bias0a = bv;
        a.M = 4096; a.N = 1024; a.K = 1024;
        gemm_kern<0,0,0,0,0,1><<<dim3(8, 32, 1), blk, 0, stream>>>(a);
    }
    {
        GemmArgs a{};
        a.A = pos_emb; a.a_rs = 1024; a.Bm = Wp; a.b_rs = 1024;
        a.out0 = pb; a.o_rs = 1024;
        a.M = 4096; a.N = 1024; a.K = 1024;
        gemm_kern<0,0,0,0,0,0><<<dim3(8, 32, 1), blk, 0, stream>>>(a);
    }
    // 5) content scores S = q_u @ k^T, batched over (b,h)
    {
        GemmArgs a{};
        a.A = qu; a.a_rs = 1024;
        a.a_bs_b = (long long)T * D; a.a_bs_h = 64;
        a.Bm = kb; a.b_rs = 1024;
        a.b_bs_b = (long long)T * D; a.b_bs_h = 64;
        a.out0 = S; a.o_rs = 1024;
        a.o_bs_b = 16 * TT; a.o_bs_h = TT;
        a.M = 1024; a.N = 1024; a.K = 64;
        gemm_kern<1,1,0,0,0,0><<<dim3(8, 8, 64), blk, 0, stream>>>(a);
    }
    // 6) unshifted pos scores X = q_v @ p^T
    {
        GemmArgs a{};
        a.A = qv; a.a_rs = 1024;
        a.a_bs_b = (long long)T * D; a.a_bs_h = 64;
        a.Bm = pb; a.b_rs = 1024;
        a.b_bs_b = (long long)T * D; a.b_bs_h = 64;
        a.out0 = X; a.o_rs = 1024;
        a.o_bs_b = 16 * TT; a.o_bs_h = TT;
        a.M = 1024; a.N = 1024; a.K = 64;
        gemm_kern<1,1,0,0,0,0><<<dim3(8, 8, 64), blk, 0, stream>>>(a);
    }
    // 7) softmax with relative shift (in-place on S)
    softmax_relshift<<<dim3(1024, 64, 1), dim3(128), 0, stream>>>(S, X);
    // 8) ctx = P @ V  (V is [K][N] in memory -> transpose while staging)
    {
        GemmArgs a{};
        a.A = S; a.a_rs = 1024;
        a.a_bs_b = 16 * TT; a.a_bs_h = TT;
        a.Bm = vb; a.b_rs = 1024;
        a.b_bs_b = (long long)T * D; a.b_bs_h = 64;
        a.out0 = ctx; a.o_rs = 1024;
        a.o_bs_b = (long long)T * D; a.o_bs_h = 64;
        a.M = 1024; a.N = 64; a.K = 1024;
        gemm_kern<1,1,1,0,0,0><<<dim3(1, 8, 64), blk, 0, stream>>>(a);
    }
    // 9) out = ctx @ Wo^T + bo  (fp32 output)
    {
        GemmArgs a{};
        a.A = ctx; a.a_rs = 1024;
        a.Bm = Wo; a.b_rs = 1024;
        a.out0 = d_out; a.o_rs = 1024;
        a.bias0a = bo;
        a.M = 4096; a.N = 1024; a.K = 1024;
        gemm_kern<1,0,0,0,1,1><<<dim3(8, 32, 1), blk, 0, stream>>>(a);
    }
}
